// LLM_Enhanced_RGCNConv_5832565588317
// MI455X (gfx1250) — matrix operand order repaired and checked by reading, not compiled, not run on hardware
//
#include <hip/hip_runtime.h>
#include <hip/hip_bf16.h>
#include <stdint.h>

#define N_NODES 100000
#define N_EDGES 250000
#define IN_CH   256
#define OUT_CH  256
#define REL_DIM 1536
#define N_REL   64
#define HIDDEN  512

typedef __attribute__((ext_vector_type(16))) __bf16 v16bf;
typedef __attribute__((ext_vector_type(8)))  float  v8f;

union FragU { v16bf v; uint4 u[2]; };

static __device__ __forceinline__ unsigned short f2bf(float f) {
  unsigned u = __builtin_bit_cast(unsigned, f);
  u += 0x7FFFu + ((u >> 16) & 1u);          // round-to-nearest-even
  return (unsigned short)(u >> 16);
}

// A fragment of 16x32 bf16 (row-major, ldk stride). ISA 16-bit A layout:
// lanes 0-15 (M=lane): K = k0..k0+7 (v0-3) and k0+16..k0+23 (v4-7); lanes 16-31: +8.
static __device__ __forceinline__ v16bf load_fragA(const unsigned short* base,
                                                   int row, int ldk, int k0, int lane) {
  const unsigned short* p = base + (size_t)row * ldk + k0 + ((lane >> 4) << 3);
  FragU f;
  f.u[0] = *(const uint4*)(p);
  f.u[1] = *(const uint4*)(p + 16);
  return f.v;
}

// B fragment of 32x16 bf16, sourced from pre-transposed Bt[N][K] (K contiguous).
// B layout (matches A): lanes 0-15 (N=lane): K = k0..k0+7 and k0+16..k0+23; lanes 16-31: +8.
static __device__ __forceinline__ v16bf load_fragB(const unsigned short* base,
                                                   int col, int ldk, int k0, int lane) {
  const unsigned short* p = base + (size_t)col * ldk + k0 + ((lane >> 4) << 3);
  FragU f;
  f.u[0] = *(const uint4*)(p);
  f.u[1] = *(const uint4*)(p + 16);
  return f.v;
}

// ---------------- conversion kernels ----------------
__global__ void cvt_f32_bf16(const float* __restrict__ src,
                             unsigned short* __restrict__ dst, int n4) {
  int i = blockIdx.x * blockDim.x + threadIdx.x;
  if (i >= n4) return;
  float4 f = ((const float4*)src)[i];
  ushort4 o;
  o.x = f2bf(f.x); o.y = f2bf(f.y); o.z = f2bf(f.z); o.w = f2bf(f.w);
  ((ushort4*)dst)[i] = o;
}

// dst[j*Kdim + k] = bf16(src[k*srcld + j])   (j < Ndim, k < Kdim)
__global__ void cvt_transpose(const float* __restrict__ src, int srcld,
                              int Kdim, int Ndim, unsigned short* __restrict__ dst) {
  int i = blockIdx.x * blockDim.x + threadIdx.x;
  if (i >= Ndim * Kdim) return;
  int j = i / Kdim, k = i % Kdim;
  dst[i] = f2bf(src[(size_t)k * srcld + j]);
}

// R[r][j] = b1[j] + sum_d rel[r][d] * W1[256+d][j]   (64 x 512, fp32, tiny)
__global__ void rel_precompute(const float* __restrict__ rel, const float* __restrict__ W1,
                               const float* __restrict__ b1, float* __restrict__ R) {
  int r = blockIdx.x;
  int j = threadIdx.x;                    // 512 threads
  float acc = b1[j];
  const float* rp = rel + (size_t)r * REL_DIM;
  const float* wp = W1 + (size_t)IN_CH * HIDDEN + j;
  for (int d = 0; d < REL_DIM; ++d)
    acc = fmaf(rp[d], wp[(size_t)d * HIDDEN], acc);
  R[(size_t)r * HIDDEN + j] = acc;
}

// ---------------- WMMA GEMM: C[M,N] = A(bf16[M,K]) x Bt(bf16[N,K])^T + bias ----------------
__global__ void gemm_bf16_wmma(const unsigned short* __restrict__ A,
                               const unsigned short* __restrict__ Bt,
                               const float* __restrict__ bias,
                               float* __restrict__ C,
                               int mtiles, int ntiles, int K) {
  int wave = (int)((blockIdx.x * blockDim.x + threadIdx.x) >> 5);
  int lane = threadIdx.x & 31;
  if (wave >= mtiles * ntiles) return;    // wave-uniform: EXEC stays all-ones
  int mt = wave / ntiles, nt = wave % ntiles;
  int col  = (nt << 4) + (lane & 15);
  int rowA = (mt << 4) + (lane & 15);
  int N = ntiles << 4;

  v8f c;
  float bv = bias ? bias[col] : 0.0f;
#pragma unroll
  for (int i = 0; i < 8; ++i) c[i] = bv;

#pragma unroll 4
  for (int k0 = 0; k0 < K; k0 += 32) {
    v16bf a = load_fragA(A, rowA, K, k0, lane);
    v16bf b = load_fragB(Bt, col, K, k0, lane);
    c = __builtin_amdgcn_wmma_f32_16x16x32_bf16(false, a, false, b, (short)0, c, false, false);
  }
  // C layout: vgpr r, lane l -> (M = r + 8*(l>=16), N = l&15)
  int rbase = (mt << 4) + ((lane >> 4) << 3);
#pragma unroll
  for (int r = 0; r < 8; ++r)
    C[(size_t)(rbase + r) * N + col] = c[r];
}

// ---------------- edge kernel: h=relu(XW1[src]+R[et]); msg = h@W2 + b2; atomic scatter ----------------
#define HSTR 520   // 512 + 8 bf16 pad -> conflict-free b128 LDS fragment loads
__global__ void edge_mlp_scatter(const int* __restrict__ ei, const int* __restrict__ et,
                                 const float* __restrict__ XW1, const float* __restrict__ R,
                                 const unsigned short* __restrict__ W2t,
                                 const float* __restrict__ b2,
                                 float* __restrict__ out_acc) {
  __shared__ unsigned short h[16 * HSTR];
  __shared__ int s_src[16], s_dst[16], s_et[16];
  int t = threadIdx.x;
  int e0 = blockIdx.x << 4;
  if (t < 16) {
    s_src[t] = ei[e0 + t];
    s_dst[t] = ei[N_EDGES + e0 + t];
    s_et[t]  = et[e0 + t];
  }
  __syncthreads();

  // build h tile [16][512] bf16 in LDS
  {
    int m  = t >> 4;
    int kc = (t & 15) << 5;              // 32 k per thread
    const float* xr = XW1 + (size_t)s_src[m] * HIDDEN + kc;
    const float* rr = R   + (size_t)s_et[m]  * HIDDEN + kc;
    unsigned short* hp = h + m * HSTR + kc;
#pragma unroll
    for (int i = 0; i < 8; ++i) {
      float4 a = ((const float4*)xr)[i];
      float4 b = ((const float4*)rr)[i];
      ushort4 o;
      o.x = f2bf(fmaxf(a.x + b.x, 0.f));
      o.y = f2bf(fmaxf(a.y + b.y, 0.f));
      o.z = f2bf(fmaxf(a.z + b.z, 0.f));
      o.w = f2bf(fmaxf(a.w + b.w, 0.f));
      ((ushort4*)hp)[i] = o;
    }
  }
  __syncthreads();

  int wave = t >> 5, lane = t & 31;
#pragma unroll
  for (int sub = 0; sub < 2; ++sub) {
    int nt  = (wave << 1) + sub;
    int col = (nt << 4) + (lane & 15);
    v8f c;
    float bv = b2[col];
#pragma unroll
    for (int i = 0; i < 8; ++i) c[i] = bv;
#pragma unroll 4
    for (int k0 = 0; k0 < HIDDEN; k0 += 32) {
      const unsigned short* pa = h + (lane & 15) * HSTR + k0 + ((lane >> 4) << 3);
      FragU fa;
      fa.u[0] = *(const uint4*)(pa);
      fa.u[1] = *(const uint4*)(pa + 16);
      v16bf b = load_fragB(W2t, col, HIDDEN, k0, lane);
      c = __builtin_amdgcn_wmma_f32_16x16x32_bf16(false, fa.v, false, b, (short)0, c, false, false);
    }
    int rb = (lane >> 4) << 3;
#pragma unroll
    for (int r = 0; r < 8; ++r)
      atomicAdd(&out_acc[(size_t)s_dst[rb + r] * OUT_CH + col], c[r]);
  }
}

// ---------------- LayerNorm + ReLU, one wave32 per node ----------------
__global__ void ln_relu(const float* __restrict__ acc, const float* __restrict__ gamma,
                        const float* __restrict__ beta, float* __restrict__ out) {
  int node = blockIdx.x * 8 + (threadIdx.x >> 5);
  int lane = threadIdx.x & 31;
  const float* p = acc + (size_t)node * OUT_CH + lane * 8;
  float4 a = ((const float4*)p)[0];
  float4 b = ((const float4*)p)[1];
  float s = a.x + a.y + a.z + a.w + b.x + b.y + b.z + b.w;
  float q = a.x*a.x + a.y*a.y + a.z*a.z + a.w*a.w + b.x*b.x + b.y*b.y + b.z*b.z + b.w*b.w;
#pragma unroll
  for (int off = 16; off; off >>= 1) {
    s += __shfl_xor(s, off, 32);
    q += __shfl_xor(q, off, 32);
  }
  float mu  = s * (1.0f / OUT_CH);
  float var = q * (1.0f / OUT_CH) - mu * mu;
  float rs  = rsqrtf(var + 1e-5f);
  float4 g0 = ((const float4*)(gamma + lane * 8))[0];
  float4 g1 = ((const float4*)(gamma + lane * 8))[1];
  float4 e0 = ((const float4*)(beta  + lane * 8))[0];
  float4 e1 = ((const float4*)(beta  + lane * 8))[1];
  float4 o0, o1;
  o0.x = fmaxf((a.x - mu) * rs * g0.x + e0.x, 0.f);
  o0.y = fmaxf((a.y - mu) * rs * g0.y + e0.y, 0.f);
  o0.z = fmaxf((a.z - mu) * rs * g0.z + e0.z, 0.f);
  o0.w = fmaxf((a.w - mu) * rs * g0.w + e0.w, 0.f);
  o1.x = fmaxf((b.x - mu) * rs * g1.x + e1.x, 0.f);
  o1.y = fmaxf((b.y - mu) * rs * g1.y + e1.y, 0.f);
  o1.z = fmaxf((b.z - mu) * rs * g1.z + e1.z, 0.f);
  o1.w = fmaxf((b.w - mu) * rs * g1.w + e1.w, 0.f);
  float* op = out + (size_t)node * OUT_CH + lane * 8;
  ((float4*)op)[0] = o0;
  ((float4*)op)[1] = o1;
}

extern "C" void kernel_launch(void* const* d_in, const int* in_sizes, int n_in,
                              void* d_out, int out_size, void* d_ws, size_t ws_size,
                              hipStream_t stream) {
  const float* x     = (const float*)d_in[0];
  const int*   ei    = (const int*)d_in[1];
  const int*   et    = (const int*)d_in[2];
  const float* rel   = (const float*)d_in[3];
  const float* W1    = (const float*)d_in[4];
  const float* b1    = (const float*)d_in[5];
  const float* W2    = (const float*)d_in[6];
  const float* b2    = (const float*)d_in[7];
  const float* Ws    = (const float*)d_in[8];
  const float* bs    = (const float*)d_in[9];
  const float* gamma = (const float*)d_in[10];
  const float* beta  = (const float*)d_in[11];
  float* out = (float*)d_out;

  uintptr_t w = (uintptr_t)d_ws;
  auto carve = [&](size_t bytes) {
    uintptr_t p = (w + 255) & ~(uintptr_t)255;
    w = p + bytes;
    return (void*)p;
  };
  unsigned short* xb   = (unsigned short*)carve((size_t)N_NODES * IN_CH * 2);   // ~51 MB
  unsigned short* w1xt = (unsigned short*)carve((size_t)HIDDEN * IN_CH * 2);
  unsigned short* w2t  = (unsigned short*)carve((size_t)OUT_CH * HIDDEN * 2);
  unsigned short* wst  = (unsigned short*)carve((size_t)OUT_CH * IN_CH * 2);
  float* Rm            = (float*)carve((size_t)N_REL * HIDDEN * 4);
  float* XW1           = (float*)carve((size_t)N_NODES * HIDDEN * 4);           // ~205 MB
  float* out_acc       = (float*)carve((size_t)N_NODES * OUT_CH * 4);           // ~102 MB

  int n4 = N_NODES * IN_CH / 4;
  cvt_f32_bf16<<<(n4 + 255) / 256, 256, 0, stream>>>(x, xb, n4);
  cvt_transpose<<<(HIDDEN * IN_CH + 255) / 256, 256, 0, stream>>>(W1, HIDDEN, IN_CH, HIDDEN, w1xt);
  cvt_transpose<<<(OUT_CH * HIDDEN + 255) / 256, 256, 0, stream>>>(W2, OUT_CH, HIDDEN, OUT_CH, w2t);
  cvt_transpose<<<(OUT_CH * IN_CH + 255) / 256, 256, 0, stream>>>(Ws, OUT_CH, IN_CH, OUT_CH, wst);
  rel_precompute<<<N_REL, HIDDEN, 0, stream>>>(rel, W1, b1, Rm);

  {   // XW1 = xb @ W1x   [100000 x 512], K=256
    int mt = N_NODES / 16, nt = HIDDEN / 16;
    long long waves = (long long)mt * nt;
    gemm_bf16_wmma<<<(int)((waves * 32 + 255) / 256), 256, 0, stream>>>(
        xb, w1xt, nullptr, XW1, mt, nt, IN_CH);
  }
  {   // out_acc = xb @ Ws + bs   [100000 x 256], K=256
    int mt = N_NODES / 16, nt = OUT_CH / 16;
    long long waves = (long long)mt * nt;
    gemm_bf16_wmma<<<(int)((waves * 32 + 255) / 256), 256, 0, stream>>>(
        xb, wst, bs, out_acc, mt, nt, IN_CH);
  }
  edge_mlp_scatter<<<N_EDGES / 16, 256, 0, stream>>>(ei, et, XW1, Rm, w2t, b2, out_acc);
  ln_relu<<<N_NODES / 8, 256, 0, stream>>>(out_acc, gamma, beta, out);
}
